// FAE_GCN_77653008712166
// MI455X (gfx1250) — matrix operand order repaired and checked by reading, not compiled, not run on hardware
//
#include <hip/hip_runtime.h>
#include <hip/hip_bf16.h>

typedef __attribute__((ext_vector_type(16))) __bf16 v16bf;
typedef __attribute__((ext_vector_type(8)))  float  v8f;

#define NNODES 100000
#define NEDGES 1600000
#define INDIM  256
#define H1DIM  64
#define H2DIM  32

// ---------------- degree / norm ----------------
__global__ void k_zero(float* p, int n) {
  int i = blockIdx.x * blockDim.x + threadIdx.x;
  if (i < n) p[i] = 0.0f;
}

__global__ void k_degree(const int* __restrict__ dst, float* __restrict__ deg) {
  int e = blockIdx.x * blockDim.x + threadIdx.x;
  if (e < NEDGES) atomicAdd(&deg[dst[e]], 1.0f);
}

__global__ void k_dinv(const float* __restrict__ deg, float* __restrict__ dinv) {
  int i = blockIdx.x * blockDim.x + threadIdx.x;
  if (i < NNODES) dinv[i] = rsqrtf(deg[i] + 1.0f);  // +1 = self loop
}

// ---------------- weight repack into WMMA B-operand order (bf16) ----------------
// B tile (32x16 bf16): lane l holds col N = l&15; elements e=0..15 hold
// K = kstep*32 + (l<16 ? 0 : 16) + e.
__global__ void k_pack_w1(const float* __restrict__ W, __bf16* __restrict__ p) {
  int i = blockIdx.x * blockDim.x + threadIdx.x;      // 4*8*32*16 = 16384
  if (i >= 4 * 8 * 32 * 16) return;
  int e = i & 15, lane = (i >> 4) & 31, ks = (i >> 9) & 7, c = i >> 12;
  int col = c * 16 + (lane & 15);
  int K   = ks * 32 + ((lane & 16) ? 16 : 0) + e;
  p[i] = (__bf16)W[K * H1DIM + col];
}

__global__ void k_pack_w2(const float* __restrict__ W, __bf16* __restrict__ p) {
  int i = blockIdx.x * blockDim.x + threadIdx.x;      // 2*2*32*16 = 2048
  if (i >= 2 * 2 * 32 * 16) return;
  int e = i & 15, lane = (i >> 4) & 31, ks = (i >> 9) & 1, c = i >> 10;
  int col = c * 16 + (lane & 15);
  int K   = ks * 32 + ((lane & 16) ? 16 : 0) + e;
  p[i] = (__bf16)W[K * H2DIM + col];
}

// ---------------- GEMM1: h1 = x @ W1 ; agg1 = b1 + dinv^2*h1 (fused epilogue) ----
__global__ __launch_bounds__(256) void k_gemm1(const float* __restrict__ x,
                                               const __bf16* __restrict__ w1p,
                                               const float* __restrict__ dinv,
                                               const float* __restrict__ b1,
                                               float* __restrict__ h1,
                                               float* __restrict__ agg1) {
  // stage packed B (32 KB) in LDS once per block
  __shared__ __bf16 sB[4 * 8 * 32 * 16];
  {
    const uint4* g = (const uint4*)w1p;
    uint4* s = (uint4*)sB;
    for (int i = threadIdx.x; i < 2048; i += 256) s[i] = g[i];
  }
  __syncthreads();

  int wave = (blockIdx.x * blockDim.x + threadIdx.x) >> 5;
  int lane = threadIdx.x & 31;
  int rowBase = wave * 16;
  if (rowBase >= NNODES) return;                      // wave-uniform, after barrier
  int row = rowBase + (lane & 15);
  int hi  = lane >> 4;
  const float* xr = x + (size_t)row * INDIM;

  auto loadA = [&](int ks) -> v16bf {
    const float* a0 = xr + ks * 32 + (hi ? 16 : 0);
    const float* a1 = xr + ks * 32 + (hi ? 24 : 8);
    v16bf a;
    #pragma unroll
    for (int e = 0; e < 8; ++e) { a[e] = (__bf16)a0[e]; a[8 + e] = (__bf16)a1[e]; }
    return a;
  };

  v8f acc[4] = {{}, {}, {}, {}};
  v16bf aCur = loadA(0);
  #pragma unroll
  for (int ks = 0; ks < 8; ++ks) {
    v16bf aNext = (ks < 7) ? loadA(ks + 1) : aCur;    // pipeline A one step ahead
    #pragma unroll
    for (int c = 0; c < 4; ++c) {
      v16bf b = *(const v16bf*)(sB + (((c * 8 + ks) * 32 + lane) << 4));
      acc[c] = __builtin_amdgcn_wmma_f32_16x16x32_bf16(
          false, aCur, false, b, (short)0, acc[c], false, false);
    }
    aCur = aNext;
  }

  // fused epilogue: write h1 and agg1 = b + dinv^2 * h1 (self-loop + bias)
  float bb[4];
  #pragma unroll
  for (int c = 0; c < 4; ++c) bb[c] = b1[c * 16 + (lane & 15)];
  #pragma unroll
  for (int r = 0; r < 8; ++r) {
    int rowM = rowBase + r + hi * 8;
    float di = dinv[rowM];
    float dd = di * di;
    #pragma unroll
    for (int c = 0; c < 4; ++c) {
      int col = c * 16 + (lane & 15);
      float v = acc[c][r];
      h1[(size_t)rowM * H1DIM + col]   = v;
      agg1[(size_t)rowM * H1DIM + col] = fmaf(dd, v, bb[c]);
    }
  }
}

// ---------------- GEMM2: h2 = relu(agg1) @ W2 ; agg2 = b2 + dinv^2*h2 ------------
__global__ __launch_bounds__(256) void k_gemm2(const float* __restrict__ agg1,
                                               const __bf16* __restrict__ w2p,
                                               const float* __restrict__ dinv,
                                               const float* __restrict__ b2,
                                               float* __restrict__ h2,
                                               float* __restrict__ agg2) {
  __shared__ __bf16 sB[2 * 2 * 32 * 16];              // 4 KB
  {
    const uint4* g = (const uint4*)w2p;
    uint4* s = (uint4*)sB;
    for (int i = threadIdx.x; i < 256; i += 256) s[i] = g[i];
  }
  __syncthreads();

  int wave = (blockIdx.x * blockDim.x + threadIdx.x) >> 5;
  int lane = threadIdx.x & 31;
  int rowBase = wave * 16;
  if (rowBase >= NNODES) return;
  int row = rowBase + (lane & 15);
  int hi  = lane >> 4;
  const float* ar = agg1 + (size_t)row * H1DIM;

  auto loadA = [&](int ks) -> v16bf {
    const float* a0 = ar + ks * 32 + (hi ? 16 : 0);
    const float* a1 = ar + ks * 32 + (hi ? 24 : 8);
    v16bf a;
    #pragma unroll
    for (int e = 0; e < 8; ++e) {
      a[e]     = (__bf16)fmaxf(a0[e], 0.0f);          // fused relu
      a[8 + e] = (__bf16)fmaxf(a1[e], 0.0f);
    }
    return a;
  };

  v8f acc[2] = {{}, {}};
  v16bf aCur = loadA(0);
  #pragma unroll
  for (int ks = 0; ks < 2; ++ks) {
    v16bf aNext = (ks < 1) ? loadA(ks + 1) : aCur;
    #pragma unroll
    for (int c = 0; c < 2; ++c) {
      v16bf b = *(const v16bf*)(sB + (((c * 2 + ks) * 32 + lane) << 4));
      acc[c] = __builtin_amdgcn_wmma_f32_16x16x32_bf16(
          false, aCur, false, b, (short)0, acc[c], false, false);
    }
    aCur = aNext;
  }

  float bb[2];
  #pragma unroll
  for (int c = 0; c < 2; ++c) bb[c] = b2[c * 16 + (lane & 15)];
  #pragma unroll
  for (int r = 0; r < 8; ++r) {
    int rowM = rowBase + r + hi * 8;
    float di = dinv[rowM];
    float dd = di * di;
    #pragma unroll
    for (int c = 0; c < 2; ++c) {
      int col = c * 16 + (lane & 15);
      float v = acc[c][r];
      h2[(size_t)rowM * H2DIM + col]   = v;
      agg2[(size_t)rowM * H2DIM + col] = fmaf(dd, v, bb[c]);
    }
  }
}

// ---------------- edge scatter: one wave per edge, lanes stride features -------
__global__ void k_edge_agg64(const int* __restrict__ src, const int* __restrict__ dst,
                             const float* __restrict__ dinv,
                             const float* __restrict__ h, float* __restrict__ agg) {
  int g = blockIdx.x * blockDim.x + threadIdx.x;
  int e = g >> 5, lane = threadIdx.x & 31;
  if (e >= NEDGES) return;
  int s = src[e], d = dst[e];
  float w = dinv[s] * dinv[d];
  float v0 = w * h[(size_t)s * 64 + lane];
  float v1 = w * h[(size_t)s * 64 + 32 + lane];
  atomicAdd(&agg[(size_t)d * 64 + lane], v0);
  atomicAdd(&agg[(size_t)d * 64 + 32 + lane], v1);
}

__global__ void k_edge_agg32(const int* __restrict__ src, const int* __restrict__ dst,
                             const float* __restrict__ dinv,
                             const float* __restrict__ h, float* __restrict__ agg) {
  int g = blockIdx.x * blockDim.x + threadIdx.x;
  int e = g >> 5, lane = threadIdx.x & 31;
  if (e >= NEDGES) return;
  int s = src[e], d = dst[e];
  float w = dinv[s] * dinv[d];
  atomicAdd(&agg[(size_t)d * 32 + lane], w * h[(size_t)s * 32 + lane]);
}

// ---------------- final linear: out = relu(agg2) @ Wlin + blin -----------------
__global__ void k_final(const float* __restrict__ agg2, const float* __restrict__ Wl,
                        const float* __restrict__ bl, float* __restrict__ out) {
  int g = blockIdx.x * blockDim.x + threadIdx.x;
  int n = g >> 5, lane = threadIdx.x & 31;
  if (n >= NNODES) return;
  float v = fmaxf(agg2[(size_t)n * 32 + lane], 0.0f) * Wl[lane];
  #pragma unroll
  for (int o = 16; o > 0; o >>= 1) v += __shfl_xor(v, o, 32);
  if (lane == 0) out[n] = v + bl[0];
}

extern "C" void kernel_launch(void* const* d_in, const int* in_sizes, int n_in,
                              void* d_out, int out_size, void* d_ws, size_t ws_size,
                              hipStream_t stream) {
  const float* x  = (const float*)d_in[0];
  const int*   ei = (const int*)  d_in[1];
  const float* W1 = (const float*)d_in[2];
  const float* b1 = (const float*)d_in[3];
  const float* W2 = (const float*)d_in[4];
  const float* b2 = (const float*)d_in[5];
  const float* Wl = (const float*)d_in[6];
  const float* bl = (const float*)d_in[7];
  float* out = (float*)d_out;

  const int* src = ei;            // edge_index[0,:]
  const int* dst = ei + NEDGES;   // edge_index[1,:]

  float* ws   = (float*)d_ws;
  float* deg  = ws;                              // N
  float* dinv = deg  + NNODES;                   // N
  float* h1   = dinv + NNODES;                   // N*64
  float* agg1 = h1   + (size_t)NNODES * 64;      // N*64
  float* h2   = agg1 + (size_t)NNODES * 64;      // N*32
  float* agg2 = h2   + (size_t)NNODES * 32;      // N*32
  __bf16* w1p = (__bf16*)(agg2 + (size_t)NNODES * 32);  // 16384 (32B aligned)
  __bf16* w2p = w1p + 16384;                            // 2048

  const int T = 256;
  // norm
  k_zero  <<<(NNODES + T - 1) / T, T, 0, stream>>>(deg, NNODES);
  k_degree<<<(NEDGES + T - 1) / T, T, 0, stream>>>(dst, deg);
  k_dinv  <<<(NNODES + T - 1) / T, T, 0, stream>>>(deg, dinv);
  // weight repack
  k_pack_w1<<<64, T, 0, stream>>>(W1, w1p);
  k_pack_w2<<< 8, T, 0, stream>>>(W2, w2p);
  // layer 1: GEMM (writes h1 + initialized agg1) then edge scatter
  int gemmBlocks = ((NNODES / 16) + 7) / 8;      // 6250 waves, 8 per block
  k_gemm1<<<gemmBlocks, T, 0, stream>>>(x, w1p, dinv, b1, h1, agg1);
  k_edge_agg64<<<(NEDGES * 32 + T - 1) / T, T, 0, stream>>>(src, dst, dinv, h1, agg1);
  // layer 2 (relu fused into gemm2 A-load)
  k_gemm2<<<gemmBlocks, T, 0, stream>>>(agg1, w2p, dinv, b2, h2, agg2);
  k_edge_agg32<<<(NEDGES * 32 + T - 1) / T, T, 0, stream>>>(src, dst, dinv, h2, agg2);
  // final linear (relu fused)
  k_final<<<(NNODES * 32 + T - 1) / T, T, 0, stream>>>(agg2, Wl, bl, out);
}
